// RPN_54056458387981
// MI455X (gfx1250) — matrix operand order repaired and checked by reading, not compiled, not run on hardware
//
#include <hip/hip_runtime.h>
#include <hip/hip_bf16.h>

// ---------------- types ----------------
typedef __bf16 bf16x16 __attribute__((ext_vector_type(16)));
typedef float  f32x8   __attribute__((ext_vector_type(8)));
typedef int    v4i_t   __attribute__((__vector_size__(16)));
typedef __attribute__((address_space(1))) v4i_t g_v4i;   // global (device) AS
typedef __attribute__((address_space(3))) v4i_t s_v4i;   // LDS (shared) AS

union FragBF { uint4 u[2]; bf16x16 v; };

#define NEGV   (-1e9f)
#define MAXD   4.135166556742356f   // log(1000/16)
#define NT     255780               // total anchors per image
#define PREVK  6000
#define POSTK  1000
#define TILES  2666                 // sum of ceil(H*W/32) over levels
#define NCHUNK 72                   // 9 taps * 8 c-chunks of 32

#if defined(__has_builtin)
#if __has_builtin(__builtin_amdgcn_global_load_async_to_lds_b128) && \
    __has_builtin(__builtin_amdgcn_s_wait_asynccnt)
#define HAVE_ASYNC_LDS 1
#endif
#endif

// ---------------- static geometry ----------------
__device__ const int   d_H[5]        = {13, 25, 50, 100, 200};
__device__ const int   d_W[5]        = {20, 40, 80, 160, 320};
__device__ const int   d_HW[5]       = {260, 1000, 4000, 16000, 64000};
__device__ const int   d_tileBase[6] = {0, 9, 41, 166, 666, 2666};
__device__ const int   d_aBase[6]    = {0, 780, 3780, 15780, 63780, 255780};
__device__ const float d_stride[5]   = {64.f, 32.f, 16.f, 8.f, 4.f};
__device__ const float d_scale[5]    = {32.f, 16.f, 8.f, 4.f, 2.f};   // BASE_SCALE * off_stride
__device__ const float d_ws0[3]      = {16.f, 11.f, 9.f};             // round(sqrt(256/ratio))
__device__ const float d_hs0[3]      = {16.f, 22.f, 27.f};            // round(ws0*ratio)

__device__ __forceinline__ unsigned short f2bf(float f) {
  unsigned u = __float_as_uint(f);
  u += 0x7FFFu + ((u >> 16) & 1u);       // round-to-nearest-even
  return (unsigned short)(u >> 16);
}

// ---------------- weight packing ----------------
// Wp2: chunk-major conv weights: [chunk][n][klocal], chunk = r*8 + cc,
//      k = r*256 + cc*32 + klocal (r = kh*3+kw). 16 KB contiguous per chunk.
// Hp[n][k]: heads, n: 0..5 cls, 6..17 reg, 18..31 zero.
__global__ void pack_kernel(const float* __restrict__ conv_w,
                            const float* __restrict__ cls_w,
                            const float* __restrict__ reg_w,
                            unsigned short* __restrict__ Wp2,
                            unsigned short* __restrict__ Hp) {
  int i = blockIdx.x * 256 + threadIdx.x;
  const int NW = 256 * 2304;
  if (i < NW) {
    int chunk = i >> 13;              // /8192
    int rem   = i & 8191;
    int n = rem >> 5, kl = rem & 31;
    int r = chunk >> 3, cc = chunk & 7;
    int c = cc * 32 + kl;
    Wp2[i] = f2bf(conv_w[n * 2304 + c * 9 + r]);   // OIHW: n*2304 + c*9 + (kh*3+kw)
  } else {
    int j = i - NW;
    if (j < 32 * 256) {
      int n = j >> 8, k = j & 255;
      float v = 0.f;
      if (n < 6)       v = cls_w[n * 256 + k];
      else if (n < 18) v = reg_w[(n - 6) * 256 + k];
      Hp[j] = f2bf(v);
    }
  }
}

// ---------------- fused conv3x3 + relu + heads (WMMA) ----------------
// One workgroup = 32 positions x 256 out-channels of one level/batch.
// 8 waves: mw = wave>>2 (16-row half), nw = wave&3 (64-col quarter).
// Double-buffered A (LDS-staged activations) and B (async-to-LDS weights).
__global__ __launch_bounds__(256)
void conv_head_kernel(const float* __restrict__ f0, const float* __restrict__ f1,
                      const float* __restrict__ f2, const float* __restrict__ f3,
                      const float* __restrict__ f4,
                      const float* __restrict__ conv_b,
                      const float* __restrict__ cls_b,
                      const float* __restrict__ reg_b,
                      const unsigned short* __restrict__ Wp2,
                      const unsigned short* __restrict__ Hp,
                      float* __restrict__ scores,
                      float* __restrict__ offs) {
  __shared__ __align__(16) unsigned short Atile[2][32 * 40];   // 32 pos x 32 K (stride 40)
  __shared__ __align__(16) unsigned short Btile[2][32 * 256];  // [n][klocal] 16 KB per buf
  __shared__ __align__(16) unsigned short Ttile[32 * 264];     // 32 pos x 256 ch
  __shared__ __align__(16) float          Hout[32 * 33];       // raw head outputs

  const int tid  = threadIdx.x;
  const int wave = tid >> 5, lane = tid & 31;
  const int laneM = lane & 15, kgrp = lane >> 4;

  int blk = blockIdx.x;
  int b = blk / TILES;
  int t = blk - b * TILES;
  int lev = 0;
#pragma unroll
  for (int l = 1; l < 5; ++l) if (t >= d_tileBase[l]) lev = l;
  int tileIn = t - d_tileBase[lev];
  int H = d_H[lev], W = d_W[lev], HW = d_HW[lev];
  const float* feat;
  if      (lev == 0) feat = f0;
  else if (lev == 1) feat = f1;
  else if (lev == 2) feat = f2;
  else if (lev == 3) feat = f3;
  else               feat = f4;
  feat += (size_t)b * 256u * (size_t)HW;

  int p0 = tileIn * 32;
  // A-fill geometry: thread = (mA, kq): position mA, 4 channels kq*4..+3 per chunk
  int mA = tid & 31, kq = tid >> 5;
  int pA = p0 + mA;
  bool pval = pA < HW;
  int hA = pval ? pA / W : 0;
  int wA = pval ? pA - hA * W : 0;

  int mw = wave >> 2, nw = wave & 3;
  int mFragOff = (16 * mw + laneM) * 80 + kgrp * 16;   // bytes into an A buffer

  // ---- staging helpers ----
  auto stageA = [&](int c, int buf) {
    int r = c >> 3, cc = c & 7;
    int y = hA + (r / 3) - 1, x = wA + (r - (r / 3) * 3) - 1;
    bool v = pval && ((unsigned)y < (unsigned)H) && ((unsigned)x < (unsigned)W);
    unsigned base = v ? (unsigned)(y * W + x) : 0u;
    const float* sp = feat + (size_t)(unsigned)(cc * 32 + kq * 4) * (size_t)HW + base;
    float v0 = sp[0];
    float v1 = sp[(size_t)HW];
    float v2 = sp[2 * (size_t)HW];
    float v3 = sp[3 * (size_t)HW];
    if (!v) { v0 = 0.f; v1 = 0.f; v2 = 0.f; v3 = 0.f; }
    uint2 pk;
    pk.x = (unsigned)f2bf(v0) | ((unsigned)f2bf(v1) << 16);
    pk.y = (unsigned)f2bf(v2) | ((unsigned)f2bf(v3) << 16);
    *(uint2*)((char*)&Atile[buf][0] + mA * 80 + kq * 8) = pk;
  };
  auto stageB = [&](int c, int buf) {
    const unsigned short* src = Wp2 + (size_t)c * 8192;
#ifdef HAVE_ASYNC_LDS
#pragma unroll
    for (int i = 0; i < 4; ++i) {
      unsigned short* g = const_cast<unsigned short*>(src) + i * 2048 + tid * 8;
      char* l = (char*)&Btile[buf][0] + i * 4096 + tid * 16;
      __builtin_amdgcn_global_load_async_to_lds_b128((g_v4i*)g, (s_v4i*)l, 0, 0);
    }
#else
#pragma unroll
    for (int i = 0; i < 4; ++i) {
      uint4 d = *(const uint4*)(src + i * 2048 + tid * 8);
      *(uint4*)((char*)&Btile[buf][0] + i * 4096 + tid * 16) = d;
    }
#endif
  };

  const f32x8 zero8 = {0.f, 0.f, 0.f, 0.f, 0.f, 0.f, 0.f, 0.f};
  f32x8 acc[4] = {zero8, zero8, zero8, zero8};

  // prologue: stage chunk 0
  stageB(0, 0);
  stageA(0, 0);
#ifdef HAVE_ASYNC_LDS
  __builtin_amdgcn_s_wait_asynccnt(0);
#endif
  __syncthreads();

  for (int c = 0; c < NCHUNK; ++c) {
    int cur = c & 1, nxt = cur ^ 1;
    if (c + 1 < NCHUNK) {            // overlap staging of next chunk with compute
      stageB(c + 1, nxt);
      stageA(c + 1, nxt);
    }
    if (c + 2 < NCHUNK)              // prefetch chunk after next into caches
      __builtin_prefetch((const void*)(Wp2 + (size_t)(c + 2) * 8192 + tid * 8), 0, 0);

    // A fragment (16-bit A 16x32 layout: kbase = (lane>=16)*8, +16 in upper 4 VGPRs)
    FragBF af;
    af.u[0] = *(const uint4*)((const char*)&Atile[cur][0] + mFragOff);
    af.u[1] = *(const uint4*)((const char*)&Atile[cur][0] + mFragOff + 32);
#pragma unroll
    for (int nt = 0; nt < 4; ++nt) {
      int n = nw * 64 + nt * 16 + laneM;
      const char* bb = (const char*)&Btile[cur][0] + n * 64 + kgrp * 16;
      FragBF bfm;
      bfm.u[0] = *(const uint4*)bb;
      bfm.u[1] = *(const uint4*)(bb + 32);
      acc[nt] = __builtin_amdgcn_wmma_f32_16x16x32_bf16(
          false, af.v, false, bfm.v, (short)0, acc[nt], false, false);
    }
#ifdef HAVE_ASYNC_LDS
    __builtin_amdgcn_s_wait_asynccnt(0);
#endif
    __syncthreads();
  }

  // bias + relu, store t tile as bf16
#pragma unroll
  for (int nt = 0; nt < 4; ++nt) {
    int n = nw * 64 + nt * 16 + laneM;
    float bias = conv_b[n];
#pragma unroll
    for (int vv = 0; vv < 8; ++vv) {
      float val = acc[nt][vv] + bias;
      val = val > 0.f ? val : 0.f;
      int m = 16 * mw + kgrp * 8 + vv;   // D layout: row = v + 8*(lane>=16)
      Ttile[m * 264 + n] = f2bf(val);
    }
  }
  __syncthreads();

  // heads as one more WMMA GEMM: [32 x 256] * [256 x 32] on waves 0..3
  if (wave < 4) {
    int mw2 = wave & 1, nt2 = wave >> 1;
    int mF = 16 * mw2 + laneM;
    int nF = 16 * nt2 + laneM;
    f32x8 hacc = zero8;
#pragma unroll
    for (int ck = 0; ck < 8; ++ck) {
      FragBF af, bfm;
      const char* ab = (const char*)Ttile + mF * 528 + ck * 64 + kgrp * 16;
      af.u[0] = *(const uint4*)ab;
      af.u[1] = *(const uint4*)(ab + 32);
      const unsigned short* hp = Hp + (size_t)nF * 256 + ck * 32 + kgrp * 8;
      bfm.u[0] = *(const uint4*)hp;
      bfm.u[1] = *(const uint4*)(hp + 16);
      hacc = __builtin_amdgcn_wmma_f32_16x16x32_bf16(
          false, af.v, false, bfm.v, (short)0, hacc, false, false);
    }
    float hb = 0.f;
    if (nF < 6)       hb = cls_b[nF];
    else if (nF < 18) hb = reg_b[nF - 6];
#pragma unroll
    for (int vv = 0; vv < 8; ++vv) {
      int m = 16 * mw2 + kgrp * 8 + vv;
      Hout[m * 33 + nF] = hacc[vv] + hb;
    }
  }
  __syncthreads();

  // softmax scores + raw offsets out
  if (tid < 96) {
    int m = tid / 3, a = tid - (tid / 3) * 3;
    int p = p0 + m;
    if (p < HW) {
      size_t idx = (size_t)b * NT + (size_t)d_aBase[lev] + (size_t)p * 3 + a;
      float c0 = Hout[m * 33 + 2 * a];
      float c1 = Hout[m * 33 + 2 * a + 1];
      scores[idx] = 1.f / (1.f + expf(c0 - c1));   // softmax(...)[1]
      float* o = offs + idx * 4;
      o[0] = Hout[m * 33 + 6 + 4 * a + 0];
      o[1] = Hout[m * 33 + 6 + 4 * a + 1];
      o[2] = Hout[m * 33 + 6 + 4 * a + 2];
      o[3] = Hout[m * 33 + 6 + 4 * a + 3];
    }
  }
}

// ---------------- init (hist/ctl/compact buffers) ----------------
__global__ void init_kernel(unsigned int* __restrict__ hist, int* __restrict__ ctl,
                            float* __restrict__ cScores) {
  int i = blockIdx.x * 256 + threadIdx.x;
  if (i < 2 * 65536) hist[i] = 0u;
  if (i < 16)        ctl[i]  = 0;
  if (i < 2 * PREVK) cScores[i] = NEGV;
}

// ---------------- decode + clip + min-box filter + score histogram ----------------
__global__ void decode_kernel(const float* __restrict__ im_info,
                              float* __restrict__ scores,
                              const float* __restrict__ offs,
                              float* __restrict__ boxes,
                              unsigned int* __restrict__ hist) {
  int i = blockIdx.x * 256 + threadIdx.x;
  if (i >= 2 * NT) return;
  int b = i / NT;
  int idx = i - b * NT;
  int lev = 0;
#pragma unroll
  for (int l = 1; l < 5; ++l) if (idx >= d_aBase[l]) lev = l;
  int li = idx - d_aBase[lev];
  int p = li / 3, a = li - (li / 3) * 3;
  int W = d_W[lev];
  int h = p / W, w = p - (p / W) * W;

  float scale = d_scale[lev];
  float aw = d_ws0[a] * scale;          // = x2-x1+1 of anchor
  float ah = d_hs0[a] * scale;
  float acx = w * d_stride[lev] + 8.f;  // x1 + 0.5*aw simplifies to shift + 8
  float acy = h * d_stride[lev] + 8.f;

  const float* o = offs + (size_t)i * 4;
  float dx = o[0], dy = o[1];
  float dw = fminf(o[2], MAXD), dh = fminf(o[3], MAXD);
  float pcx = dx * aw + acx, pcy = dy * ah + acy;
  float pw = aw * expf(dw), ph = ah * expf(dh);

  float xm = im_info[b * 6 + 1] - 1.f;
  float ym = im_info[b * 6 + 0] - 1.f;
  float x1 = fminf(fmaxf(pcx - 0.5f * pw, 0.f), xm);
  float x2 = fminf(fmaxf(pcx + 0.5f * pw, 0.f), xm);
  float y1 = fminf(fmaxf(pcy - 0.5f * ph, 0.f), ym);
  float y2 = fminf(fmaxf(pcy + 0.5f * ph, 0.f), ym);
  float* bx = boxes + (size_t)i * 4;
  bx[0] = x1; bx[1] = y1; bx[2] = x2; bx[3] = y2;

  float thr = 2.f * im_info[b * 6 + 2];
  float s = scores[i];
  if (!((x2 - x1 + 1.f) > thr && (y2 - y1 + 1.f) > thr)) { s = NEGV; scores[i] = s; }

  unsigned u = __float_as_uint(s);
  unsigned key = (u & 0x80000000u) ? ~u : (u | 0x80000000u);   // order-preserving
  atomicAdd(&hist[b * 65536 + (key >> 16)], 1u);
}

// ---------------- find top-6000 threshold bucket ----------------
__global__ void scan_kernel(const unsigned int* __restrict__ hist, int* __restrict__ ctl) {
  __shared__ unsigned int csum[256];
  __shared__ unsigned int suffix[256];
  int b = blockIdx.x, tid = threadIdx.x;
  const unsigned int* hb = hist + b * 65536;
  unsigned int s = 0;
  for (int j = 0; j < 256; ++j) s += hb[tid * 256 + j];
  csum[tid] = s;
  __syncthreads();
  if (tid == 0) {
    unsigned int run = 0;
    for (int t = 255; t >= 0; --t) { suffix[t] = run; run += csum[t]; }
    ctl[b * 8 + 0] = -1;      // T (default: take everything)
    ctl[b * 8 + 1] = 0;       // count strictly above T
    ctl[b * 8 + 2] = 0;       // slots for tie bucket
  }
  __syncthreads();
  unsigned int run = suffix[tid];
  for (int j = 255; j >= 0; --j) {
    int bk = tid * 256 + j;
    unsigned int c = hb[bk];
    if (run < PREVK && run + c >= PREVK) {   // unique finder
      ctl[b * 8 + 0] = bk;
      ctl[b * 8 + 1] = (int)run;
      ctl[b * 8 + 2] = PREVK - (int)run;
    }
    run += c;
  }
}

// ---------------- compact top-6000 candidates ----------------
__global__ void compact_kernel(const float* __restrict__ scores,
                               const float* __restrict__ boxes,
                               int* __restrict__ ctl,
                               float* __restrict__ cScores,
                               float* __restrict__ cBoxes) {
  int i = blockIdx.x * 256 + threadIdx.x;
  if (i >= 2 * NT) return;
  int b = i / NT;
  float s = scores[i];
  unsigned u = __float_as_uint(s);
  unsigned key = (u & 0x80000000u) ? ~u : (u | 0x80000000u);
  int bk = (int)(key >> 16);
  int T = ctl[b * 8 + 0], above = ctl[b * 8 + 1], tie = ctl[b * 8 + 2];
  int slot = -1;
  if (bk > T) {
    slot = atomicAdd(&ctl[b * 8 + 3], 1);
  } else if (bk == T) {
    int ts = atomicAdd(&ctl[b * 8 + 4], 1);
    if (ts < tie) slot = above + ts;
  }
  if (slot >= 0 && slot < PREVK) {
    size_t d = (size_t)b * PREVK + slot;
    cScores[d] = s;
    const float* bx = boxes + (size_t)i * 4;
    float* cb = cBoxes + d * 4;
    cb[0] = bx[0]; cb[1] = bx[1]; cb[2] = bx[2]; cb[3] = bx[3];
  }
}

// ---------------- greedy NMS, whole candidate set LDS-resident ----------------
__global__ __launch_bounds__(1024)
void nms_kernel(const float* __restrict__ cScores,
                const float* __restrict__ cBoxes,
                float* __restrict__ out) {
  extern __shared__ float sm[];
  float* sw   = sm;              // 6000
  float* bx1  = sm + 6000;
  float* by1  = sm + 12000;
  float* bx2  = sm + 18000;
  float* by2  = sm + 24000;
  float* area = sm + 30000;
  float* rv   = sm + 36000;      // 1024
  int*   ri   = (int*)(sm + 37024);
  float* sel  = sm + 38048;      // 8

  int b = blockIdx.x, tid = threadIdx.x;
  for (int j = tid; j < PREVK; j += 1024) {
    size_t d = (size_t)b * PREVK + j;
    float s = cScores[d];
    const float* cb = cBoxes + d * 4;
    float X1 = cb[0], Y1 = cb[1], X2 = cb[2], Y2 = cb[3];
    sw[j] = s; bx1[j] = X1; by1[j] = Y1; bx2[j] = X2; by2[j] = Y2;
    area[j] = (X2 - X1 + 1.f) * (Y2 - Y1 + 1.f);
  }
  __syncthreads();

  for (int it = 0; it < POSTK; ++it) {
    float bv = -3.4e38f; int bi = 0;
    for (int j = tid; j < PREVK; j += 1024) {
      float s = sw[j];
      if (s > bv) { bv = s; bi = j; }
    }
    rv[tid] = bv; ri[tid] = bi;
    __syncthreads();
    for (int off = 512; off > 0; off >>= 1) {
      if (tid < off) {
        float ov = rv[tid + off]; int oi = ri[tid + off];
        if (ov > rv[tid] || (ov == rv[tid] && oi < ri[tid])) { rv[tid] = ov; ri[tid] = oi; }
      }
      __syncthreads();
    }
    if (tid == 0) {
      int i = ri[0];
      sel[0] = bx1[i]; sel[1] = by1[i]; sel[2] = bx2[i]; sel[3] = by2[i];
      sel[4] = rv[0];  sel[5] = area[i];
    }
    __syncthreads();
    float sx1 = sel[0], sy1 = sel[1], sx2 = sel[2], sy2 = sel[3];
    float si = sel[4], sA = sel[5];
    for (int j = tid; j < PREVK; j += 1024) {
      float ix1 = fmaxf(sx1, bx1[j]), iy1 = fmaxf(sy1, by1[j]);
      float ix2 = fminf(sx2, bx2[j]), iy2 = fminf(sy2, by2[j]);
      float inter = fmaxf(ix2 - ix1 + 1.f, 0.f) * fmaxf(iy2 - iy1 + 1.f, 0.f);
      float iou = inter / (sA + area[j] - inter);
      if (iou > 0.7f) sw[j] = NEGV;   // suppresses the selected box itself too
    }
    if (tid == 0) {
      size_t ro = ((size_t)b * POSTK + it) * 5;
      bool val = si > (NEGV * 0.5f);
      out[ro + 0] = val ? (float)b : 0.f;
      out[ro + 1] = val ? sx1 : 0.f;
      out[ro + 2] = val ? sy1 : 0.f;
      out[ro + 3] = val ? sx2 : 0.f;
      out[ro + 4] = val ? sy2 : 0.f;
    }
    __syncthreads();
  }
}

// ---------------- host-side orchestration ----------------
extern "C" void kernel_launch(void* const* d_in, const int* in_sizes, int n_in,
                              void* d_out, int out_size, void* d_ws, size_t ws_size,
                              hipStream_t stream) {
  (void)in_sizes; (void)n_in; (void)out_size; (void)ws_size;
  const float* f0      = (const float*)d_in[0];
  const float* f1      = (const float*)d_in[1];
  const float* f2      = (const float*)d_in[2];
  const float* f3      = (const float*)d_in[3];
  const float* f4      = (const float*)d_in[4];
  const float* im_info = (const float*)d_in[5];
  const float* conv_w  = (const float*)d_in[6];
  const float* conv_b  = (const float*)d_in[7];
  const float* cls_w   = (const float*)d_in[8];
  const float* cls_b   = (const float*)d_in[9];
  const float* reg_w   = (const float*)d_in[10];
  const float* reg_b   = (const float*)d_in[11];

  char* ws = (char*)d_ws;
  size_t off = 0;
  auto alloc = [&](size_t bytes) {
    char* p = ws + off;
    off += (bytes + 255) & ~(size_t)255;
    return p;
  };
  float* scores          = (float*)alloc((size_t)2 * NT * 4);
  float* offs            = (float*)alloc((size_t)2 * NT * 4 * 4);
  float* boxes           = (float*)alloc((size_t)2 * NT * 4 * 4);
  unsigned short* Wp2    = (unsigned short*)alloc((size_t)256 * 2304 * 2);
  unsigned short* Hp     = (unsigned short*)alloc((size_t)32 * 256 * 2);
  unsigned int*   hist   = (unsigned int*)alloc((size_t)2 * 65536 * 4);
  int*            ctl    = (int*)alloc((size_t)16 * 4);
  float* cScores         = (float*)alloc((size_t)2 * PREVK * 4);
  float* cBoxes          = (float*)alloc((size_t)2 * PREVK * 16);

  pack_kernel<<<(256 * 2304 + 32 * 256 + 255) / 256, 256, 0, stream>>>(
      conv_w, cls_w, reg_w, Wp2, Hp);
  conv_head_kernel<<<2 * TILES, 256, 0, stream>>>(
      f0, f1, f2, f3, f4, conv_b, cls_b, reg_b, Wp2, Hp, scores, offs);
  init_kernel<<<(2 * 65536 + 255) / 256, 256, 0, stream>>>(hist, ctl, cScores);
  decode_kernel<<<(2 * NT + 255) / 256, 256, 0, stream>>>(
      im_info, scores, offs, boxes, hist);
  scan_kernel<<<2, 256, 0, stream>>>(hist, ctl);
  compact_kernel<<<(2 * NT + 255) / 256, 256, 0, stream>>>(
      scores, boxes, ctl, cScores, cBoxes);
  nms_kernel<<<2, 1024, (size_t)(38056 + 8) * sizeof(float), stream>>>(
      cScores, cBoxes, (float*)d_out);
}
